// LlamaAttention_29609504539091
// MI455X (gfx1250) — matrix operand order repaired and checked by reading, not compiled, not run on hardware
//
#include <hip/hip_runtime.h>

// ---------------------------------------------------------------------------
// LlamaAttention forward on gfx1250 (MI455X): bf16 WMMA everywhere, f32 softmax
// B=2, S=2048, D=1024, H=16, HD=64
// Adds Tensor Data Mover (TDM) staging of GEMM A-tiles into LDS.
// ---------------------------------------------------------------------------
#define BB 2
#define SS 2048
#define DD 1024
#define HH 16
#define HDD 64
#define NN (BB * SS) // 4096 rows

typedef __attribute__((ext_vector_type(16))) __bf16 v16bf;
typedef __attribute__((ext_vector_type(8)))  float  v8f;

union FragBF {
    v16bf v;
    uint4 q[2];
    unsigned short u[16];
};

__device__ __forceinline__ unsigned short f32_to_bf16_rne(float f) {
    unsigned u = __float_as_uint(f);
    unsigned r = (u + 0x7FFFu + ((u >> 16) & 1u)) >> 16;
    return (unsigned short)r;
}

__device__ __forceinline__ v8f wmma_bf16(const FragBF& a, const FragBF& b, v8f c) {
    return __builtin_amdgcn_wmma_f32_16x16x32_bf16(
        /*neg_a=*/false, a.v, /*neg_b=*/false, b.v,
        /*c_mod=*/(short)0, c, /*reuse_a=*/false, /*reuse_b=*/false);
}

// ---------------------------------------------------------------------------
// TDM: stage a 16 x 1024 bf16 row-major tile from global into LDS with one
// tensor_load_to_lds (D# per CDNA5 ISA).
// ---------------------------------------------------------------------------
#if __has_builtin(__builtin_amdgcn_tensor_load_to_lds)
#define HAVE_TDM 1
typedef unsigned int tdm_v4u __attribute__((ext_vector_type(4)));
typedef int          tdm_v8i __attribute__((ext_vector_type(8)));
typedef int          tdm_v4i __attribute__((ext_vector_type(4)));

__device__ __forceinline__ void tdm_load_tile_16xD_bf16(
    const unsigned short* gsrc, unsigned short* lds_dst) {
    const unsigned long long ga = (unsigned long long)gsrc;
    // Flat shared address truncated to 32 bits == LDS byte offset (ISA §10.2).
    const unsigned lds = (unsigned)(unsigned long long)lds_dst;

    tdm_v4u g0;
    g0.x = 1u;                                            // count=1, user mode
    g0.y = lds;                                           // lds_addr
    g0.z = (unsigned)(ga & 0xFFFFFFFFull);                // global_addr[31:0]
    g0.w = (unsigned)((ga >> 32) & 0x01FFFFFFull)         // global_addr[56:32]
           | (2u << 30);                                  // type = 2 ("image")

    // group1 bit-packing (bits noted are within the 256-bit group):
    //  [17:16]=data_size(2B)  [79:48]=tensor_dim0=1024  [111:80]=tensor_dim1=16
    //  [127:112]=tile_dim0=1024  [143:128]=tile_dim1=16  [207:160]=dim0_stride=1024
    tdm_v8i g1 = { 0x00010000,
                   (int)(DD << 16),
                   (int)(16 << 16),
                   (int)(DD << 16),
                   16,
                   DD,
                   0,
                   0 };
    tdm_v4i z4 = {0, 0, 0, 0};
#if defined(__clang_major__) && (__clang_major__ >= 23)
    tdm_v8i z8 = {0, 0, 0, 0, 0, 0, 0, 0};
    __builtin_amdgcn_tensor_load_to_lds(g0, g1, z4, z4, z8, 0);
#else
    __builtin_amdgcn_tensor_load_to_lds(g0, g1, z4, z4, 0);
#endif
}
#endif // HAVE_TDM

// Stage one 16x1024 bf16 tile into LDS (TDM if available, else vector copy).
// Must be called by all 128 threads; includes the trailing barrier.
__device__ __forceinline__ void stage_tile_16xD(const unsigned short* gsrc,
                                                unsigned short* xt,
                                                int tid, int wave) {
#if HAVE_TDM
    if (wave == 0) {
        tdm_load_tile_16xD_bf16(gsrc, xt);
        __builtin_amdgcn_s_wait_tensorcnt(0);
    }
    __syncthreads();
#else
    const uint4* xg = (const uint4*)gsrc;
    uint4* xl = (uint4*)xt;
#pragma unroll
    for (int i = 0; i < 16; ++i) xl[tid + 128 * i] = xg[tid + 128 * i];
    __syncthreads();
#endif
}

// ---------------------------------------------------------------------------
// Kernel 0: fp32 -> bf16 (raw ushort) conversion
// ---------------------------------------------------------------------------
__global__ void cvt_f32_to_bf16_kernel(const float* __restrict__ src,
                                       unsigned short* __restrict__ dst, int n) {
    int i = blockIdx.x * blockDim.x + threadIdx.x;
    if (i < n) dst[i] = f32_to_bf16_rne(src[i]);
}

// ---------------------------------------------------------------------------
// Kernel 1: fused QKV projection (+RoPE for Q,K; transpose-store for V)
// grid = (256 row tiles, 4 head groups, 3 {Q,K,V}), block = 128 (4 waves)
// Each wave: one 16x64 output tile == one full head for 16 tokens.
// ---------------------------------------------------------------------------
__global__ void qkv_rope_kernel(const unsigned short* __restrict__ Xb,
                                const unsigned short* __restrict__ Wqb,
                                const unsigned short* __restrict__ Wkb,
                                const unsigned short* __restrict__ Wvb,
                                const float* __restrict__ cosT,
                                const float* __restrict__ sinT,
                                unsigned short* __restrict__ Qb,
                                unsigned short* __restrict__ Kb,
                                unsigned short* __restrict__ Vt) {
    __shared__ unsigned short xt[16 * DD]; // 32 KB: X tile 16x1024 bf16

    const int tid  = threadIdx.x;
    const int lane = tid & 31;
    const int wave = tid >> 5;
    const int hl   = lane >> 4;   // half of wave
    const int nl   = lane & 15;   // lane-in-half
    const int rowTile = blockIdx.x;            // 0..255
    const int head    = blockIdx.y * 4 + wave; // 0..15
    const int which   = blockIdx.z;            // 0=Q 1=K 2=V
    const int rowBase = rowTile * 16;

    stage_tile_16xD(Xb + (long)rowBase * DD, xt, tid, wave);

    const unsigned short* W = (which == 0) ? Wqb : (which == 1) ? Wkb : Wvb;

    v8f acc[4] = {};
    for (int kb = 0; kb < DD; kb += 32) {
        // A fragment from LDS: row m = nl, K = kb + hl*16 + e
        FragBF a;
        a.q[0] = *(const uint4*)(xt + nl * DD + kb + hl * 16);
        a.q[1] = *(const uint4*)(xt + nl * DD + kb + 8 + hl * 16);
        __builtin_prefetch(W + (head * HDD + nl) * DD + kb + 64, 0, 0);
#pragma unroll
        for (int nt = 0; nt < 4; ++nt) {
            const int col = head * HDD + nt * 16 + nl; // output col == W row
            FragBF b;
            const uint4* wp = (const uint4*)(W + (long)col * DD + kb + hl * 16);
            b.q[0] = wp[0];
            b.q[1] = wp[1];
            acc[nt] = wmma_bf16(a, b, acc[nt]);
        }
    }

    const int bI    = rowBase / SS;
    const int sBase = rowBase % SS;

    if (which == 2) {
        // V: store transposed  Vt[((b*H+h)*64 + c) * S + s]
#pragma unroll
        for (int nt = 0; nt < 4; ++nt)
#pragma unroll
            for (int r = 0; r < 8; ++r) {
                const int m = r + 8 * hl;
                const int s = sBase + m;
                const int c = nt * 16 + nl;
                Vt[(((long)bI * HH + head) * HDD + c) * SS + s] =
                    f32_to_bf16_rne(acc[nt][r]);
            }
    } else {
        unsigned short* dst = (which == 0) ? Qb : Kb;
#pragma unroll
        for (int nt = 0; nt < 4; ++nt)
#pragma unroll
            for (int r = 0; r < 8; ++r) {
                const int m = r + 8 * hl;
                const int s = sBase + m;
                const int c = nt * 16 + nl;
                const float x       = acc[nt][r];
                const float partner = acc[nt ^ 2][r]; // c +/- 32 lives 2 tiles away
                const float cs = cosT[s * HDD + c];
                const float sn = sinT[s * HDD + c];
                const float rot = (nt < 2) ? -partner : partner; // rotate_half
                dst[(((long)bI * HH + head) * SS + s) * HDD + c] =
                    f32_to_bf16_rne(x * cs + rot * sn);
            }
    }
}

// ---------------------------------------------------------------------------
// Kernel 2: attention core. grid = (128 q-tiles, 16 heads, 2 batch),
// block = 256 (8 waves). Dynamic LDS: scores f32[16][2048] (128K) +
// P bf16[16][2048] (64K) + Opart f32[8][16][64] (32K) = 224 KB (<320 KB WGP).
// ---------------------------------------------------------------------------
__global__ void attn_kernel(const unsigned short* __restrict__ Qb,
                            const unsigned short* __restrict__ Kb,
                            const unsigned short* __restrict__ Vt,
                            const unsigned char* __restrict__ mask,
                            unsigned short* __restrict__ Ob) {
    extern __shared__ unsigned char smem[];
    float*          sc    = (float*)smem;                               // 16x2048 f32
    unsigned short* P     = (unsigned short*)(smem + 16 * SS * 4);      // 16x2048 bf16
    float*          Opart = (float*)(smem + 16 * SS * 4 + 16 * SS * 2); // 8x16x64 f32

    const int tid  = threadIdx.x;
    const int lane = tid & 31;
    const int wave = tid >> 5;
    const int hl   = lane >> 4;
    const int nl   = lane & 15;
    const int qt = blockIdx.x, h = blockIdx.y, b = blockIdx.z;
    const int qBase = qt * 16;
    const long bh = (long)b * HH + h;
    const unsigned short* Qp = Qb + (bh * SS + qBase) * HDD;
    const unsigned short* Kp = Kb + bh * SS * HDD;
    const unsigned short* Vp = Vt + bh * HDD * SS;

    // Q A-fragments for HD=64 (two k32 chunks), kept in registers
    FragBF qa[2];
#pragma unroll
    for (int c = 0; c < 2; ++c) {
        qa[c].q[0] = *(const uint4*)(Qp + nl * HDD + c * 32 + hl * 16);
        qa[c].q[1] = *(const uint4*)(Qp + nl * HDD + c * 32 + 8 + hl * 16);
    }

    // ---- scores: each wave owns 256 kv columns (16 tiles of 16) ----
    const float scale = 0.125f; // HD^-0.5
    for (int j = 0; j < 16; ++j) {
        const int kv = wave * 256 + j * 16;
        v8f accS = {};
#pragma unroll
        for (int c = 0; c < 2; ++c) {
            FragBF kf; // B = K^T : col nl -> key row kv+nl, K-dim contiguous
            const uint4* kp =
                (const uint4*)(Kp + (long)(kv + nl) * HDD + c * 32 + hl * 16);
            kf.q[0] = kp[0];
            kf.q[1] = kp[1];
            accS = wmma_bf16(qa[c], kf, accS);
        }
        const float mk = mask[b * SS + kv + nl] ? 0.0f : -3.0e38f;
#pragma unroll
        for (int r = 0; r < 8; ++r) {
            const int m = r + 8 * hl;
            sc[m * SS + kv + nl] = accS[r] * scale + mk;
        }
    }
    __syncthreads();

    // ---- softmax: 16 threads per row, full 2048-wide rows in f32 ----
    {
        const int row = tid >> 4, l16 = tid & 15;
        float* srow = sc + row * SS;
        float mx = -3.4e38f;
        for (int i = l16; i < SS; i += 16) mx = fmaxf(mx, srow[i]);
#pragma unroll
        for (int o = 8; o > 0; o >>= 1) mx = fmaxf(mx, __shfl_xor(mx, o, 16));
        float sum = 0.0f;
        for (int i = l16; i < SS; i += 16) {
            const float e = __expf(srow[i] - mx);
            srow[i] = e;
            sum += e;
        }
#pragma unroll
        for (int o = 8; o > 0; o >>= 1) sum += __shfl_xor(sum, o, 16);
        const float inv = 1.0f / sum;
        unsigned short* prow = P + row * SS;
        for (int i = l16; i < SS; i += 16)
            prow[i] = f32_to_bf16_rne(srow[i] * inv);
    }
    __syncthreads();

    // ---- O = P @ V : each wave reduces its 256 kv columns ----
    v8f accO[4] = {};
    for (int j = 0; j < 8; ++j) {
        const int kb = wave * 256 + j * 32;
        FragBF pa; // A from LDS P
        pa.q[0] = *(const uint4*)(P + nl * SS + kb + hl * 16);
        pa.q[1] = *(const uint4*)(P + nl * SS + kb + 8 + hl * 16);
#pragma unroll
        for (int nt = 0; nt < 4; ++nt) {
            FragBF vf; // B from V^T: row = hd col, contiguous along S
            const uint4* vp =
                (const uint4*)(Vp + (long)(nt * 16 + nl) * SS + kb + hl * 16);
            vf.q[0] = vp[0];
            vf.q[1] = vp[1];
            accO[nt] = wmma_bf16(pa, vf, accO[nt]);
        }
    }
#pragma unroll
    for (int nt = 0; nt < 4; ++nt)
#pragma unroll
        for (int r = 0; r < 8; ++r) {
            const int m = r + 8 * hl;
            Opart[(wave * 16 + m) * HDD + nt * 16 + nl] = accO[nt][r];
        }
    __syncthreads();

    // cross-wave reduction + store Ob in [B,S,H*HD] row-major (GEMM-ready)
    for (int idx = tid; idx < 16 * HDD; idx += 256) {
        const int m = idx >> 6, c = idx & 63;
        float s = 0.0f;
#pragma unroll
        for (int w2 = 0; w2 < 8; ++w2) s += Opart[(w2 * 16 + m) * HDD + c];
        Ob[((long)b * SS + qBase + m) * DD + h * HDD + c] = f32_to_bf16_rne(s);
    }
}

// ---------------------------------------------------------------------------
// Kernel 3: output projection  out = O @ Wo^T  (f32 result)
// grid = (256 row tiles, 4 col groups), block = 128 (4 waves; 64 cols/wave)
// ---------------------------------------------------------------------------
__global__ void out_proj_kernel(const unsigned short* __restrict__ Ob,
                                const unsigned short* __restrict__ Wob,
                                float* __restrict__ out) {
    __shared__ unsigned short xt[16 * DD];

    const int tid  = threadIdx.x;
    const int lane = tid & 31;
    const int wave = tid >> 5;
    const int hl   = lane >> 4;
    const int nl   = lane & 15;
    const int rowTile = blockIdx.x;
    const int colTile = blockIdx.y * 4 + wave; // 0..15 -> 64-col region
    const int rowBase = rowTile * 16;

    stage_tile_16xD(Ob + (long)rowBase * DD, xt, tid, wave);

    v8f acc[4] = {};
    for (int kb = 0; kb < DD; kb += 32) {
        FragBF a;
        a.q[0] = *(const uint4*)(xt + nl * DD + kb + hl * 16);
        a.q[1] = *(const uint4*)(xt + nl * DD + kb + 8 + hl * 16);
        __builtin_prefetch(Wob + (colTile * 64 + nl) * DD + kb + 64, 0, 0);
#pragma unroll
        for (int nt = 0; nt < 4; ++nt) {
            const int col = colTile * 64 + nt * 16 + nl;
            FragBF b;
            const uint4* wp = (const uint4*)(Wob + (long)col * DD + kb + hl * 16);
            b.q[0] = wp[0];
            b.q[1] = wp[1];
            acc[nt] = wmma_bf16(a, b, acc[nt]);
        }
    }
#pragma unroll
    for (int nt = 0; nt < 4; ++nt)
#pragma unroll
        for (int r = 0; r < 8; ++r) {
            const int m = r + 8 * hl;
            out[(long)(rowBase + m) * DD + colTile * 64 + nt * 16 + nl] = acc[nt][r];
        }
}

// ---------------------------------------------------------------------------
// Host-side launch
// ---------------------------------------------------------------------------
extern "C" void kernel_launch(void* const* d_in, const int* in_sizes, int n_in,
                              void* d_out, int out_size, void* d_ws, size_t ws_size,
                              hipStream_t stream) {
    const float*         X    = (const float*)d_in[0];
    const unsigned char* mask = (const unsigned char*)d_in[1]; // bool [B,S]
    const float*         cosT = (const float*)d_in[2];          // [1,1,S,HD]
    const float*         sinT = (const float*)d_in[3];
    const float*         Wq   = (const float*)d_in[4];
    const float*         Wk   = (const float*)d_in[5];
    const float*         Wv   = (const float*)d_in[6];
    const float*         Wo   = (const float*)d_in[7];
    float*               out  = (float*)d_out;

    // Workspace layout (bf16 as raw ushort), ~48 MiB total:
    unsigned char* ws = (unsigned char*)d_ws;
    const size_t SZ_X = (size_t)NN * DD * 2;  // 8 MiB
    const size_t SZ_W = (size_t)DD * DD * 2;  // 2 MiB each
    unsigned short* Xb  = (unsigned short*)(ws);
    unsigned short* Wqb = (unsigned short*)(ws + SZ_X);
    unsigned short* Wkb = (unsigned short*)(ws + SZ_X + SZ_W);
    unsigned short* Wvb = (unsigned short*)(ws + SZ_X + 2 * SZ_W);
    unsigned short* Wob = (unsigned short*)(ws + SZ_X + 3 * SZ_W);
    unsigned short* Qb  = (unsigned short*)(ws + SZ_X + 4 * SZ_W);     // [B,H,S,HD]
    unsigned short* Kb  = (unsigned short*)(ws + 2 * SZ_X + 4 * SZ_W); // [B,H,S,HD]
    unsigned short* Vt  = (unsigned short*)(ws + 3 * SZ_X + 4 * SZ_W); // [B,H,HD,S]
    unsigned short* Ob  = (unsigned short*)(ws + 4 * SZ_X + 4 * SZ_W); // [B,S,D]

    // 0) fp32 -> bf16 conversions
    cvt_f32_to_bf16_kernel<<<(NN * DD + 255) / 256, 256, 0, stream>>>(X, Xb, NN * DD);
    cvt_f32_to_bf16_kernel<<<(DD * DD + 255) / 256, 256, 0, stream>>>(Wq, Wqb, DD * DD);
    cvt_f32_to_bf16_kernel<<<(DD * DD + 255) / 256, 256, 0, stream>>>(Wk, Wkb, DD * DD);
    cvt_f32_to_bf16_kernel<<<(DD * DD + 255) / 256, 256, 0, stream>>>(Wv, Wvb, DD * DD);
    cvt_f32_to_bf16_kernel<<<(DD * DD + 255) / 256, 256, 0, stream>>>(Wo, Wob, DD * DD);

    // 1) QKV projections + RoPE (+ V transpose)
    qkv_rope_kernel<<<dim3(NN / 16, HH / 4, 3), 128, 0, stream>>>(
        Xb, Wqb, Wkb, Wvb, cosT, sinT, Qb, Kb, Vt);

    // 2) attention core: 224 KB dynamic LDS per workgroup
    const size_t attn_lds = (size_t)16 * SS * 4 + (size_t)16 * SS * 2 + 8 * 16 * HDD * 4;
    attn_kernel<<<dim3(SS / 16, HH, BB), 256, attn_lds, stream>>>(Qb, Kb, Vt, mask, Ob);

    // 3) output projection -> f32 d_out
    out_proj_kernel<<<dim3(NN / 16, 4), 128, 0, stream>>>(Ob, Wob, out);
}
